// GRADE_38431367364935
// MI455X (gfx1250) — matrix operand order repaired and checked by reading, not compiled, not run on hardware
//
#include <hip/hip_runtime.h>
#include <hip/hip_bf16.h>

// ---------------------------------------------------------------------------
// UDAGCN forward loss on MI455X (gfx1250).
// Dense GEMMs -> v_wmma_f32_16x16x32_bf16 (f32 accumulate), A tiles staged via
// the Tensor Data Mover (tensor_load_to_lds, TENSORcnt double buffering).
// Sparse aggregation -> float4 gather + global_atomic_add_f32 scatter.
// ---------------------------------------------------------------------------

typedef __attribute__((ext_vector_type(16))) __bf16 bf16x16;
typedef __attribute__((ext_vector_type(8)))  float  floatx8;
typedef __attribute__((ext_vector_type(4)))  unsigned int uint32x4;
typedef __attribute__((ext_vector_type(8)))  int int32x8;
typedef __attribute__((ext_vector_type(4)))  int int32x4;

union Frag32B { uint4 q[2]; bf16x16 v; };

__device__ __forceinline__ unsigned short f2bf(float x) {
    unsigned u = __float_as_uint(x);
    unsigned r = u + 0x7FFFu + ((u >> 16) & 1u);   // round-to-nearest-even
    return (unsigned short)(r >> 16);
}

// ---------------- misc elementwise kernels ----------------

__global__ void k_zero_f32(float* __restrict__ p, long n) {
    long t = (long)blockIdx.x * blockDim.x + threadIdx.x;
    if (t < n) p[t] = 0.0f;
}

__global__ void k_degree(const int* __restrict__ idx, float* __restrict__ deg, int E) {
    int t = blockIdx.x * blockDim.x + threadIdx.x;
    if (t < E) atomicAdd(&deg[idx[t]], 1.0f);
}

__global__ void k_deg_to_norm(float* __restrict__ p, int n) {
    int t = blockIdx.x * blockDim.x + threadIdx.x;
    if (t < n) p[t] = rsqrtf(fmaxf(p[t], 1.0f));
}

// Pack W [K x Nc] f32 row-major into bf16 WMMA B-fragment order:
// Bp[((kt*NT + nt)*32 + lane)*16 + j] = bf16(W[kt*32 + (lane>=16?16:0) + j][nt*16 + lane%16])
__global__ void k_pack_w(const float* __restrict__ W, unsigned short* __restrict__ Bp,
                         int K, int Nc) {
    int idx = blockIdx.x * blockDim.x + threadIdx.x;
    if (idx >= K * Nc) return;
    int j    = idx & 15;
    int lane = (idx >> 4) & 31;
    int t    = idx >> 9;
    int NT   = Nc >> 4;
    int nt   = t % NT;
    int kt   = t / NT;
    int n    = nt * 16 + (lane & 15);
    int k    = kt * 32 + ((lane >> 4) << 4) + j;
    Bp[idx] = f2bf(W[(size_t)k * Nc + n]);
}

// out[n,k] = bf16(x[n,k] * norm[n]); processed as float4 groups.
__global__ void k_scale_rows_bf16(const float* __restrict__ x, const float* __restrict__ norm,
                                  unsigned short* __restrict__ out, int N, int K) {
    int q = K >> 2;
    long t = (long)blockIdx.x * blockDim.x + threadIdx.x;
    if (t >= (long)N * q) return;
    int n = (int)(t / q);
    float s = norm[n];
    float4 v = ((const float4*)x)[t];
    ushort4 o;
    o.x = f2bf(v.x * s); o.y = f2bf(v.y * s);
    o.z = f2bf(v.z * s); o.w = f2bf(v.w * s);
    ((ushort4*)out)[t] = o;
}

// ---------------- WMMA bf16 GEMM: C[M x Nc] = A[M x K] * B[K x Nc] ----------------
// A: bf16 row-major.  Bp: bf16 pre-packed fragments (L2-resident).  C: f32 row-major.
// Block = 256 threads (8 waves); block tile 128x128; K-step 32.
// Wave grid 2x4; each wave -> 64 rows x 32 cols = 4x2 WMMA tiles.
// A tile staged by the Tensor Data Mover, double buffered on TENSORcnt.

#define BM 128
#define BN 128
#define BK 32
#define LDA 40   // padded LDS row stride (elements): 64B row + 16B pad (TDM pad fields)

#if __has_builtin(__builtin_amdgcn_tensor_load_to_lds)
#define USE_TDM 1
#else
#define USE_TDM 0
#endif

#if USE_TDM
// Issue one TDM load of a (rows x 32) bf16 tile -> LDS, padded 16B per 64B row.
// tensor_dim1 = rows remaining in the matrix: hardware zero-fills past-the-end rows.
__device__ __forceinline__ void tdm_load_a_tile(const unsigned short* gsrc,
                                                unsigned ldsAddr,
                                                unsigned tensor_d0, unsigned tensor_d1,
                                                unsigned stride0) {
    unsigned long long ga = (unsigned long long)(uintptr_t)gsrc;
    uint32x4 g0;
    g0[0] = 1u;                                                   // count=1, user desc
    g0[1] = ldsAddr;                                              // lds_addr
    g0[2] = (unsigned)(ga & 0xffffffffu);                         // global_addr[31:0]
    g0[3] = (unsigned)((ga >> 32) & 0x01ffffffu) | (2u << 30);    // addr[56:32], type=2
    int32x8 g1;
    g1[0] = (int)((1u << 16)     // data_size = 2 bytes
                | (1u << 20)     // pad_enable
                | (3u << 22)     // pad_interval: pad after 16 DWORDs (64B)
                | (3u << 25));   // pad_amount: 4 DWORDs (16B)
    g1[1] = (int)((tensor_d0 & 0xffffu) << 16);                   // tensor_dim0[15:0]
    g1[2] = (int)((tensor_d0 >> 16) | ((tensor_d1 & 0xffffu) << 16));
    g1[3] = (int)((tensor_d1 >> 16) | ((unsigned)BK << 16));      // tile_dim0 = 32
    g1[4] = (int)((unsigned)BM);                                  // tile_dim1 = 128, tile_dim2 = 0
    g1[5] = (int)stride0;                                         // tensor_dim0_stride[31:0]
    g1[6] = 0;
    g1[7] = 0;
    int32x4 z4 = {0, 0, 0, 0};
    int32x8 z8 = {0, 0, 0, 0, 0, 0, 0, 0};
    __builtin_amdgcn_tensor_load_to_lds(g0, g1, z4, z4, z8, 0);
}
#endif

__global__ __launch_bounds__(256)
void k_gemm_bf16_wmma(const unsigned short* __restrict__ A,
                      const unsigned short* __restrict__ Bp,
                      float* __restrict__ C,
                      int M, int K, int Nc) {
    __shared__ unsigned short ldsA[2][BM * LDA];   // 2 x 10 KB

    const int tid  = threadIdx.x;
    const int wave = tid >> 5;
    const int lane = tid & 31;
    const int blockRow = blockIdx.y * BM;
    const int blockCol = blockIdx.x * BN;
    const int waveRow  = (wave >> 2) * 64;      // 0 or 64
    const int waveCol  = (wave & 3) * 32;       // 0,32,64,96

    floatx8 acc[4][2];
    const floatx8 z = {0.f,0.f,0.f,0.f,0.f,0.f,0.f,0.f};
#pragma unroll
    for (int i = 0; i < 4; ++i)
#pragma unroll
        for (int j = 0; j < 2; ++j) acc[i][j] = z;

    const int kTiles = K / BK;
    const int NT = Nc >> 4;
    const int halfSel  = (lane >> 4) & 1;   // A-fragment K-half select
    const int rowInTl  = lane & 15;

#if USE_TDM
    const unsigned ldsBase[2] = { (unsigned)(uintptr_t)&ldsA[0][0],
                                  (unsigned)(uintptr_t)&ldsA[1][0] };
    const unsigned short* gRow = A + (size_t)blockRow * K;
    const unsigned rowsLeft = (unsigned)(M - blockRow);
    if (wave == 0)
        tdm_load_a_tile(gRow, ldsBase[0], (unsigned)K, rowsLeft, (unsigned)K);
#endif

    int buf = 0;
    for (int kt = 0; kt < kTiles; ++kt) {
#if USE_TDM
        if (wave == 0) {
            if (kt + 1 < kTiles) {
                tdm_load_a_tile(gRow + (size_t)(kt + 1) * BK, ldsBase[buf ^ 1],
                                (unsigned)(K - (kt + 1) * BK), rowsLeft, (unsigned)K);
                __builtin_amdgcn_s_wait_tensorcnt(1);   // tile kt has landed
            } else {
                __builtin_amdgcn_s_wait_tensorcnt(0);
            }
        }
        __syncthreads();
#else
        // fallback: per-thread staging of A tile (128 x 32 bf16) into LDS
#pragma unroll
        for (int c = 0; c < 2; ++c) {
            int chunk = tid + c * 256;
            int r     = chunk >> 2;
            int kcol  = (chunk & 3) * 8;
            int gr    = blockRow + r; if (gr >= M) gr = M - 1;
            uint4 v = *(const uint4*)(A + (size_t)gr * K + kt * BK + kcol);
            *(uint4*)(&ldsA[buf][r * LDA + kcol]) = v;
        }
        __syncthreads();
#endif

        // --- B fragments straight from packed global (lives in L2) ---
        Frag32B bfrag[2];
#pragma unroll
        for (int j = 0; j < 2; ++j) {
            int nTile = ((blockCol + waveCol) >> 4) + j;
            const unsigned short* p = Bp + (((size_t)kt * NT + nTile) * 32 + lane) * 16;
            bfrag[j].q[0] = *(const uint4*)p;
            bfrag[j].q[1] = *(const uint4*)(p + 8);
        }

        // --- A fragments from LDS (contiguous k = 16*half + s, same order as Bp) ---
        Frag32B afrag[4];
#pragma unroll
        for (int i = 0; i < 4; ++i) {
            int r = waveRow + i * 16 + rowInTl;
            const unsigned short* p = &ldsA[buf][r * LDA + halfSel * 16];
            afrag[i].q[0] = *(const uint4*)p;          // K {0..7} or {16..23}
            afrag[i].q[1] = *(const uint4*)(p + 8);    // K {8..15} or {24..31}
        }

#pragma unroll
        for (int i = 0; i < 4; ++i)
#pragma unroll
            for (int j = 0; j < 2; ++j)
                acc[i][j] = __builtin_amdgcn_wmma_f32_16x16x32_bf16(
                    false, afrag[i].v, false, bfrag[j].v,
                    (short)0, acc[i][j], false, false);
        __syncthreads();   // all readers done before TDM overwrites this buffer
        buf ^= 1;
    }

    // --- store: C/D layout: VGPR r -> M = r (+8 for lanes 16..31), N = lane%16 ---
    const int colBase = blockCol + waveCol + (lane & 15);
    const int hiOff   = (lane >> 4) << 3;
#pragma unroll
    for (int i = 0; i < 4; ++i) {
        int rBase = blockRow + waveRow + i * 16 + hiOff;
#pragma unroll
        for (int j = 0; j < 2; ++j)
#pragma unroll
            for (int r = 0; r < 8; ++r) {
                int row = rBase + r;
                if (row < M) C[(size_t)row * Nc + colBase + j * 16] = acc[i][j][r];
            }
    }
}

// ---------------- sparse aggregation: agg[dst] += h[src] ----------------

__global__ void k_spmm_scatter(const float* __restrict__ h, const int* __restrict__ src,
                               const int* __restrict__ dst, float* __restrict__ agg,
                               int E, int Hq) {
    long t = (long)blockIdx.x * blockDim.x + threadIdx.x;
    if (t >= (long)E * Hq) return;
    int e = (int)(t / Hq);
    int g = (int)(t % Hq);
    int s = src[e], d = dst[e];
    float4 v = ((const float4*)(h + (size_t)s * Hq * 4))[g];
    float* o = agg + (size_t)d * Hq * 4 + (size_t)g * 4;
    atomicAdd(o + 0, v.x); atomicAdd(o + 1, v.y);
    atomicAdd(o + 2, v.z); atomicAdd(o + 3, v.w);
}

__global__ void k_epilogue(const float* __restrict__ agg, const float* __restrict__ norm_in,
                           const float* __restrict__ bias, float* __restrict__ out,
                           int N, int H) {
    long t = (long)blockIdx.x * blockDim.x + threadIdx.x;
    if (t >= (long)N * H) return;
    int n = (int)(t / H);
    int c = (int)(t % H);
    float v = agg[t] * norm_in[n] + bias[c];
    out[t] = v > 0.0f ? v : 0.0f;
}

// ---------------- heads & losses ----------------

// wave-per-row fc: logits[n, 0..9] = h[n,:] @ Wfc + bfc
__global__ __launch_bounds__(256)
void k_fc_head(const float* __restrict__ h, const float* __restrict__ Wfc,
               const float* __restrict__ bfc, float* __restrict__ logits, int N) {
    __shared__ float w[256 * 10];
    for (int i = threadIdx.x; i < 2560; i += 256) w[i] = Wfc[i];
    __syncthreads();
    int wave = threadIdx.x >> 5, lane = threadIdx.x & 31;
    int n = blockIdx.x * 8 + wave;
    if (n >= N) return;
    float acc[10];
#pragma unroll
    for (int c = 0; c < 10; ++c) acc[c] = 0.0f;
    for (int k = lane; k < 256; k += 32) {
        float x = h[(size_t)n * 256 + k];
#pragma unroll
        for (int c = 0; c < 10; ++c) acc[c] += x * w[k * 10 + c];
    }
#pragma unroll
    for (int c = 0; c < 10; ++c)
#pragma unroll
        for (int o = 16; o > 0; o >>= 1) acc[c] += __shfl_xor(acc[c], o, 32);
    if (lane == 0)
#pragma unroll
        for (int c = 0; c < 10; ++c) logits[(size_t)n * 10 + c] = acc[c] + bfc[c];
}

__global__ void k_class_loss(const float* __restrict__ logits, const int* __restrict__ labels,
                             float* __restrict__ acc, int N) {
    int n = blockIdx.x * blockDim.x + threadIdx.x;
    if (n >= N) return;
    const float* l = logits + (size_t)n * 10;
    float m = l[0];
#pragma unroll
    for (int c = 1; c < 10; ++c) m = fmaxf(m, l[c]);
    float s = 0.0f;
#pragma unroll
    for (int c = 0; c < 10; ++c) s += expf(l[c] - m);
    float loss = (m + logf(s)) - l[labels[n]];
    atomicAdd(acc, loss);
}

// wave-per-row domain discriminator on concat(h1, h2, logits)  (522 feats -> 2)
__global__ __launch_bounds__(256)
void k_domain_loss(const float* __restrict__ h1, const float* __restrict__ h2,
                   const float* __restrict__ lg, const float* __restrict__ Wd,
                   const float* __restrict__ bd, int dlabel,
                   float* __restrict__ acc, int N) {
    int wave = threadIdx.x >> 5, lane = threadIdx.x & 31;
    int n = blockIdx.x * 8 + wave;
    if (n >= N) return;
    float a0 = 0.0f, a1 = 0.0f;
    for (int k = lane; k < 522; k += 32) {
        float f = (k < 256) ? h1[(size_t)n * 256 + k]
                : (k < 512) ? h2[(size_t)n * 256 + (k - 256)]
                            : lg[(size_t)n * 10 + (k - 512)];
        a0 += f * Wd[k * 2 + 0];
        a1 += f * Wd[k * 2 + 1];
    }
#pragma unroll
    for (int o = 16; o > 0; o >>= 1) {
        a0 += __shfl_xor(a0, o, 32);
        a1 += __shfl_xor(a1, o, 32);
    }
    if (lane == 0) {
        float z0 = a0 + bd[0], z1 = a1 + bd[1];
        float m = fmaxf(z0, z1);
        float lse = m + logf(expf(z0 - m) + expf(z1 - m));
        float loss = lse - (dlabel == 0 ? z0 : z1);
        atomicAdd(acc, loss);
    }
}

__global__ void k_finalize(const float* __restrict__ accs, float* __restrict__ out,
                           float invN, float inv2N) {
    if (threadIdx.x == 0 && blockIdx.x == 0)
        out[0] = accs[0] * invN + 0.01f * (accs[1] * inv2N);
}

// ---------------------------------------------------------------------------

extern "C" void kernel_launch(void* const* d_in, const int* in_sizes, int n_in,
                              void* d_out, int out_size, void* d_ws, size_t ws_size,
                              hipStream_t stream) {
    (void)n_in; (void)out_size; (void)ws_size; (void)in_sizes;
    const int N = 50000, E = 800000, IN = 512, H = 256;

    const float* features_s = (const float*)d_in[0];
    const int*   labels_s   = (const int*)  d_in[1];
    const float* features_t = (const float*)d_in[2];
    const int*   src_s      = (const int*)  d_in[3];
    const int*   dst_s      = (const int*)  d_in[4];
    const int*   src_t      = (const int*)  d_in[5];
    const int*   dst_t      = (const int*)  d_in[6];
    const float* W0  = (const float*)d_in[7];
    const float* b0  = (const float*)d_in[8];
    const float* W1  = (const float*)d_in[9];
    const float* b1  = (const float*)d_in[10];
    const float* Wfc = (const float*)d_in[11];
    const float* bfc = (const float*)d_in[12];
    const float* Wd  = (const float*)d_in[13];
    const float* bd  = (const float*)d_in[14];
    float* out = (float*)d_out;

    // ---- carve workspace ----
    char* ws = (char*)d_ws;
    size_t off = 0;
    auto take = [&](size_t bytes) -> char* {
        char* p = ws + off;
        off = (off + bytes + 255) & ~(size_t)255;
        return p;
    };
    float* norms = (float*)take((size_t)4 * N * 4);   // [src_s | dst_s | src_t | dst_t]
    float* norm_src_s = norms + 0 * N;
    float* norm_dst_s = norms + 1 * N;
    float* norm_src_t = norms + 2 * N;
    float* norm_dst_t = norms + 3 * N;
    float* accs = (float*)take(256);                  // [class_sum, domain_sum]
    unsigned short* W0p  = (unsigned short*)take((size_t)IN * H * 2);
    unsigned short* W1p  = (unsigned short*)take((size_t)H  * H * 2);
    unsigned short* Abuf = (unsigned short*)take((size_t)N * IN * 2);
    float* hbuf = (float*)take((size_t)N * H * 4);
    float* agg  = (float*)take((size_t)N * H * 4);
    float* hs1  = (float*)take((size_t)N * H * 4);
    float* ht1  = (float*)take((size_t)N * H * 4);
    float* hs2  = (float*)take((size_t)N * H * 4);
    float* ht2  = (float*)take((size_t)N * H * 4);
    float* logits_s = (float*)take((size_t)N * 10 * 4);
    float* logits_t = (float*)take((size_t)N * 10 * 4);

    const int TB = 256;
    auto blocks = [](long n, int tb) { return (int)((n + tb - 1) / tb); };

    // ---- degrees -> norms ----
    k_zero_f32<<<blocks(4L * N, TB), TB, 0, stream>>>(norms, 4L * N);
    k_zero_f32<<<1, 64, 0, stream>>>(accs, 64);
    k_degree<<<blocks(E, TB), TB, 0, stream>>>(src_s, norm_src_s, E);
    k_degree<<<blocks(E, TB), TB, 0, stream>>>(dst_s, norm_dst_s, E);
    k_degree<<<blocks(E, TB), TB, 0, stream>>>(src_t, norm_src_t, E);
    k_degree<<<blocks(E, TB), TB, 0, stream>>>(dst_t, norm_dst_t, E);
    k_deg_to_norm<<<blocks(4L * N, TB), TB, 0, stream>>>(norms, 4 * N);

    // ---- pack weights to bf16 fragment order ----
    k_pack_w<<<blocks((long)IN * H, TB), TB, 0, stream>>>(W0, W0p, IN, H);
    k_pack_w<<<blocks((long)H * H, TB), TB, 0, stream>>>(W1, W1p, H, H);

    dim3 gemmGrid0(H / BN, (N + BM - 1) / BM);

    // ---- GraphConv layer 0 / layer 1 for both graphs ----
    struct Pass {
        const float* x; int K;
        const unsigned short* Wp;
        const float* bias;
        const int *src, *dst;
        const float *n_src, *n_dst;
        float* outh;
    } passes[4] = {
        { features_s, IN, W0p, b0, src_s, dst_s, norm_src_s, norm_dst_s, hs1 },
        { features_t, IN, W0p, b0, src_t, dst_t, norm_src_t, norm_dst_t, ht1 },
        { hs1,        H,  W1p, b1, src_s, dst_s, norm_src_s, norm_dst_s, hs2 },
        { ht1,        H,  W1p, b1, src_t, dst_t, norm_src_t, norm_dst_t, ht2 },
    };
    for (int p = 0; p < 4; ++p) {
        const Pass& P = passes[p];
        k_scale_rows_bf16<<<blocks((long)N * (P.K / 4), TB), TB, 0, stream>>>(
            P.x, P.n_src, Abuf, N, P.K);
        k_gemm_bf16_wmma<<<gemmGrid0, 256, 0, stream>>>(Abuf, P.Wp, hbuf, N, P.K, H);
        k_zero_f32<<<blocks((long)N * H, TB), TB, 0, stream>>>(agg, (long)N * H);
        k_spmm_scatter<<<blocks((long)E * (H / 4), TB), TB, 0, stream>>>(
            hbuf, P.src, P.dst, agg, E, H / 4);
        k_epilogue<<<blocks((long)N * H, TB), TB, 0, stream>>>(
            agg, P.n_dst, P.bias, P.outh, N, H);
    }

    // ---- heads & losses ----
    k_fc_head<<<blocks(N, 8), 256, 0, stream>>>(hs2, Wfc, bfc, logits_s, N);
    k_fc_head<<<blocks(N, 8), 256, 0, stream>>>(ht2, Wfc, bfc, logits_t, N);
    k_class_loss<<<blocks(N, TB), TB, 0, stream>>>(logits_s, labels_s, &accs[0], N);
    k_domain_loss<<<blocks(N, 8), 256, 0, stream>>>(hs1, hs2, logits_s, Wd, bd, 0, &accs[1], N);
    k_domain_loss<<<blocks(N, 8), 256, 0, stream>>>(ht1, ht2, logits_t, Wd, bd, 1, &accs[1], N);
    k_finalize<<<1, 32, 0, stream>>>(accs, out, 1.0f / N, 1.0f / (2.0f * N));
}
